// BaseAttention_47124381172257
// MI455X (gfx1250) — matrix operand order repaired and checked by reading, not compiled, not run on hardware
//
#include <hip/hip_runtime.h>
#include <hip/hip_bf16.h>

typedef __attribute__((ext_vector_type(16))) _Float16 v16h;
typedef __attribute__((ext_vector_type(8)))  _Float16 v8h;
typedef __attribute__((ext_vector_type(8)))  float    v8f;

static constexpr int Bn   = 4;
static constexpr int S_LEN = 2048;
static constexpr int Dd   = 256;
static constexpr int Hn   = 8;
static constexpr int KDim = 256;          // per-head dim (Keras key_dim = D)
static constexpr int BS   = Bn * S_LEN;   // 8192
static constexpr int HKD  = Hn * KDim;    // 2048

// ---------------- WMMA helpers (CDNA5 layouts, wave32) ----------------

__device__ __forceinline__ v8f wmma_f16(v16h a, v16h b, v8f c) {
  // (neg_a, A, neg_b, B, c_mod, C, reuse_a, reuse_b)
  return __builtin_amdgcn_wmma_f32_16x16x32_f16(false, a, false, b, (short)0, c,
                                                false, false);
}

// A fragment 16x32 f16: lane = row (lane&15); halves 0-7 -> K = hi*8+0..7,
// halves 8-15 -> K = 16 + hi*8 + 0..7  (hi = lane>>4). src row-major, ld elems.
__device__ __forceinline__ v16h load_a(const _Float16* __restrict__ src, int ld, int lane) {
  const _Float16* p = src + (size_t)(lane & 15) * ld + ((lane >> 4) << 3);
  union { v16h v; v8h h[2]; } u;
  u.h[0] = *(const v8h*)p;
  u.h[1] = *(const v8h*)(p + 16);
  return u.v;
}

// B fragment 32x16 f16: lane = column (lane&15); halves 0-7 -> K = hi*8+0..7,
// halves 8-15 -> K = 16 + hi*8 + 0..7, from a K-contiguous (transposed, n-major) layout with leading dim ld.
__device__ __forceinline__ v16h load_b(const _Float16* __restrict__ srcT, int ld, int lane) {
  const _Float16* p = srcT + (size_t)(lane & 15) * ld + ((lane >> 4) << 3);
  union { v16h v; v8h h[2]; } u;
  u.h[0] = *(const v8h*)p;
  u.h[1] = *(const v8h*)(p + 16);
  return u.v;
}

// ---------------- Kernel 1: LayerNorm + f16 cast (3 streams) ----------------

__global__ void ln_cast_kernel(const float* __restrict__ xq, const float* __restrict__ xk,
                               const float* __restrict__ xv,
                               const float* __restrict__ gq, const float* __restrict__ bq,
                               const float* __restrict__ gk, const float* __restrict__ bk,
                               const float* __restrict__ gv, const float* __restrict__ bv,
                               _Float16* __restrict__ Xq, _Float16* __restrict__ Xk,
                               _Float16* __restrict__ Xv) {
  int token = blockIdx.x;
  int strm  = blockIdx.y;
  int d     = threadIdx.x;
  const float* x; const float* g; const float* be; _Float16* o;
  if (strm == 0)      { x = xq; g = gq; be = bq; o = Xq; }
  else if (strm == 1) { x = xk; g = gk; be = bk; o = Xk; }
  else                { x = xv; g = gv; be = bv; o = Xv; }

  float val = x[(size_t)token * Dd + d];
  __shared__ float s1[256];
  __shared__ float s2[256];
  s1[d] = val; s2[d] = val * val;
  __syncthreads();
  for (int off = 128; off > 0; off >>= 1) {
    if (d < off) { s1[d] += s1[d + off]; s2[d] += s2[d + off]; }
    __syncthreads();
  }
  float mean = s1[0] * (1.0f / Dd);
  float var  = s2[0] * (1.0f / Dd) - mean * mean;
  float rs   = rsqrtf(var + 1e-5f);
  o[(size_t)token * Dd + d] = (_Float16)(((val - mean) * rs) * g[d] + be[d]);
}

// ------------- Kernel 2: transpose + cast weights to f16 (n-major) -------------

__global__ void tcast_kernel(const float* __restrict__ in, _Float16* __restrict__ out,
                             int R, int C) {
  size_t idx = (size_t)blockIdx.x * blockDim.x + threadIdx.x;
  if (idx >= (size_t)R * C) return;
  int r = (int)(idx / C);
  int c = (int)(idx % C);
  out[(size_t)c * R + r] = (_Float16)in[idx];
}

// ------------- Kernel 3: WMMA GEMM for Q/K/V projections -------------
// Y[m,n] = sum_k A[m,k]*Wt[n,k] ; then (+bias)*qscale ; store f16 per mode:
//   mode 0: out[((b*H+h)*S + s)*KD + kd]      (Q, K head-major)
//   mode 1: out[((b*H+h)*KD + kd)*S + s]      (V transposed for P*V B-fragments)

__global__ void __launch_bounds__(128) gemm_qkv_kernel(
    const _Float16* __restrict__ A, const _Float16* __restrict__ Bt,
    const float* __restrict__ bias, _Float16* __restrict__ out,
    int Kdim, float qscale, int mode) {
  int lane = threadIdx.x & 31;
  int wave = threadIdx.x >> 5;
  int m0 = blockIdx.x * 64 + wave * 16;
  int n0 = blockIdx.y * 64;

  v8f z = {};
  v8f acc[4] = {z, z, z, z};
  for (int k0 = 0; k0 < Kdim; k0 += 32) {
    v16h a = load_a(A + (size_t)m0 * Kdim + k0, Kdim, lane);
    for (int j = 0; j < 4; ++j) {
      v16h bf = load_b(Bt + (size_t)(n0 + j * 16) * Kdim + k0, Kdim, lane);
      acc[j] = wmma_f16(a, bf, acc[j]);
    }
  }
  int hi = lane >> 4, lo = lane & 15;
  for (int j = 0; j < 4; ++j) {
    int n = n0 + j * 16 + lo;
    float bv = bias[n];
    int hh = n / KDim, kd = n % KDim;
    for (int r = 0; r < 8; ++r) {
      int m = m0 + r + 8 * hi;
      int bb = m / S_LEN, s = m % S_LEN;
      float val = (acc[j][r] + bv) * qscale;
      size_t idx;
      if (mode == 0) idx = (((size_t)(bb * Hn + hh)) * S_LEN + s) * KDim + kd;
      else           idx = (((size_t)(bb * Hn + hh)) * KDim + kd) * S_LEN + s;
      out[idx] = (_Float16)val;
    }
  }
}

// ------------- Kernel 4: flash attention, one wave per 16 queries -------------

__global__ void __launch_bounds__(32) attn_kernel(
    const _Float16* __restrict__ Q, const _Float16* __restrict__ K,
    const _Float16* __restrict__ Vt, _Float16* __restrict__ O) {
  int lane = threadIdx.x & 31;
  int qm = blockIdx.x * 16;
  int h  = blockIdx.y;
  int b  = blockIdx.z;
  size_t bh = (size_t)(b * Hn + h);
  const _Float16* Qb = Q  + bh * (size_t)S_LEN * KDim + (size_t)qm * KDim;
  const _Float16* Kb = K  + bh * (size_t)S_LEN * KDim;
  const _Float16* Vb = Vt + bh * (size_t)KDim * S_LEN;

  __shared__ _Float16 pbuf[16 * 32];

  // Preload Q fragments for all of KD (8 x 16x32 tiles)
  v16h qf[8];
  for (int kk = 0; kk < 8; ++kk) qf[kk] = load_a(Qb + kk * 32, KDim, lane);

  float mrow[8], lrow[8];
  v8f z = {};
  v8f acc[16];
  for (int r = 0; r < 8; ++r) { mrow[r] = -3.0e38f; lrow[r] = 0.0f; }
  for (int t = 0; t < 16; ++t) acc[t] = z;

  int hi = lane >> 4, lo = lane & 15;

  for (int kc = 0; kc < S_LEN; kc += 32) {
    // Prefetch next chunk's K (16KB contiguous) and V (256 rows, 4KB stride)
    // while this chunk's WMMAs execute (gfx1250 global_prefetch_b8 path).
    if (kc + 32 < S_LEN) {
      const _Float16* nk = Kb + (size_t)(kc + 32) * KDim;
      for (int pfi = 0; pfi < 4; ++pfi)
        __builtin_prefetch(nk + (size_t)lane * 256 + pfi * 64, 0, 0);
      const _Float16* nv = Vb + (kc + 32);
      for (int pfi = 0; pfi < 8; ++pfi)
        __builtin_prefetch(nv + ((size_t)(pfi * 32 + lane)) * S_LEN, 0, 0);
    }

    // scores for 32 keys = two 16x16 D tiles, K-reduction over KD
    v8f s0 = z, s1 = z;
    for (int kk = 0; kk < 8; ++kk) {
      v16h b0 = load_b(Kb + (size_t)kc * KDim + kk * 32, KDim, lane);
      v16h b1 = load_b(Kb + (size_t)(kc + 16) * KDim + kk * 32, KDim, lane);
      s0 = wmma_f16(qf[kk], b0, s0);
      s1 = wmma_f16(qf[kk], b1, s1);
    }
    // online softmax (row reductions across the 16 lanes of each half-wave)
    float p0[8], p1[8], scl[8];
    for (int r = 0; r < 8; ++r) {
      float v = fmaxf(s0[r], s1[r]);
      v = fmaxf(v, __shfl_xor(v, 1, 32));
      v = fmaxf(v, __shfl_xor(v, 2, 32));
      v = fmaxf(v, __shfl_xor(v, 4, 32));
      v = fmaxf(v, __shfl_xor(v, 8, 32));
      float mnew = fmaxf(mrow[r], v);
      scl[r] = __expf(mrow[r] - mnew);
      mrow[r] = mnew;
      p0[r] = __expf(s0[r] - mnew);
      p1[r] = __expf(s1[r] - mnew);
      float sum = p0[r] + p1[r];
      sum += __shfl_xor(sum, 1, 32);
      sum += __shfl_xor(sum, 2, 32);
      sum += __shfl_xor(sum, 4, 32);
      sum += __shfl_xor(sum, 8, 32);
      lrow[r] = lrow[r] * scl[r] + sum;
    }
    for (int t = 0; t < 16; ++t)
      for (int r = 0; r < 8; ++r) acc[t][r] *= scl[r];

    // re-lay P (D layout) into an A fragment (16x32) via LDS
    __syncthreads();
    for (int r = 0; r < 8; ++r) {
      pbuf[(r + 8 * hi) * 32 + lo]      = (_Float16)p0[r];
      pbuf[(r + 8 * hi) * 32 + 16 + lo] = (_Float16)p1[r];
    }
    __syncthreads();
    v16h pa;
    {
      union { v16h v; v8h h[2]; } u;
      const _Float16* pp = &pbuf[lo * 32 + hi * 8];
      u.h[0] = *(const v8h*)pp;
      u.h[1] = *(const v8h*)(pp + 16);
      pa = u.v;
    }
    // P (16x32) x V (32xKD), V stored transposed so B fragments are contiguous
    for (int t = 0; t < 16; ++t) {
      v16h vb = load_b(Vb + (size_t)(t * 16) * S_LEN + kc, S_LEN, lane);
      acc[t] = wmma_f16(pa, vb, acc[t]);
    }
  }

  float inv[8];
  for (int r = 0; r < 8; ++r) inv[r] = 1.0f / lrow[r];
  for (int t = 0; t < 16; ++t) {
    for (int r = 0; r < 8; ++r) {
      int row = qm + r + 8 * hi;
      int kd  = t * 16 + lo;
      O[(((size_t)b * S_LEN + row) * Hn + h) * KDim + kd] =
          (_Float16)(acc[t][r] * inv[r]);
    }
  }
}

// ------------- Kernel 5: output projection + bias + residual (f32 out) -------------

__global__ void __launch_bounds__(128) gemm_out_kernel(
    const _Float16* __restrict__ A, const _Float16* __restrict__ Bt,
    const float* __restrict__ bias, const float* __restrict__ resid,
    float* __restrict__ out, int Kdim) {
  int lane = threadIdx.x & 31;
  int wave = threadIdx.x >> 5;
  int m0 = blockIdx.x * 64 + wave * 16;
  int n0 = blockIdx.y * 64;

  v8f z = {};
  v8f acc[4] = {z, z, z, z};
  for (int k0 = 0; k0 < Kdim; k0 += 32) {
    v16h a = load_a(A + (size_t)m0 * Kdim + k0, Kdim, lane);
    for (int j = 0; j < 4; ++j) {
      v16h bf = load_b(Bt + (size_t)(n0 + j * 16) * Kdim + k0, Kdim, lane);
      acc[j] = wmma_f16(a, bf, acc[j]);
    }
  }
  int hi = lane >> 4, lo = lane & 15;
  for (int j = 0; j < 4; ++j) {
    int n = n0 + j * 16 + lo;
    float bv = bias[n];
    for (int r = 0; r < 8; ++r) {
      int m = m0 + r + 8 * hi;
      size_t idx = (size_t)m * Dd + n;
      out[idx] = resid[idx] + acc[j][r] + bv;
    }
  }
}

// ---------------- Host launcher ----------------

extern "C" void kernel_launch(void* const* d_in, const int* in_sizes, int n_in,
                              void* d_out, int out_size, void* d_ws, size_t ws_size,
                              hipStream_t stream) {
  const float* inq  = (const float*)d_in[0];
  const float* ink  = (const float*)d_in[1];
  const float* inv  = (const float*)d_in[2];
  const float* gq   = (const float*)d_in[3];
  const float* bqln = (const float*)d_in[4];
  const float* gk   = (const float*)d_in[5];
  const float* bkln = (const float*)d_in[6];
  const float* gv   = (const float*)d_in[7];
  const float* bvln = (const float*)d_in[8];
  const float* Wq   = (const float*)d_in[9];
  const float* bq   = (const float*)d_in[10];
  const float* Wk   = (const float*)d_in[11];
  const float* bk   = (const float*)d_in[12];
  const float* Wv   = (const float*)d_in[13];
  const float* bv   = (const float*)d_in[14];
  const float* Wo   = (const float*)d_in[15];
  const float* bo   = (const float*)d_in[16];
  float* out = (float*)d_out;

  _Float16* ws = (_Float16*)d_ws;
  size_t off = 0;
  _Float16* Xq  = ws + off; off += (size_t)BS * Dd;
  _Float16* Xk  = ws + off; off += (size_t)BS * Dd;
  _Float16* Xv  = ws + off; off += (size_t)BS * Dd;
  _Float16* WqT = ws + off; off += (size_t)HKD * Dd;
  _Float16* WkT = ws + off; off += (size_t)HKD * Dd;
  _Float16* WvT = ws + off; off += (size_t)HKD * Dd;
  _Float16* WoT = ws + off; off += (size_t)Dd * HKD;
  _Float16* Qh  = ws + off; off += (size_t)Bn * Hn * S_LEN * KDim;
  _Float16* Kh  = ws + off; off += (size_t)Bn * Hn * S_LEN * KDim;
  _Float16* Vht = ws + off; off += (size_t)Bn * Hn * KDim * S_LEN;
  _Float16* At  = ws + off; off += (size_t)BS * HKD;

  // 1. LayerNorm + cast (3 streams)
  ln_cast_kernel<<<dim3(BS, 3), 256, 0, stream>>>(
      inq, ink, inv, gq, bqln, gk, bkln, gv, bvln, Xq, Xk, Xv);

  // 2. Weight transpose-casts
  int wElems = Dd * HKD;
  int tgrid = (wElems + 255) / 256;
  tcast_kernel<<<tgrid, 256, 0, stream>>>(Wq, WqT, Dd, HKD);
  tcast_kernel<<<tgrid, 256, 0, stream>>>(Wk, WkT, Dd, HKD);
  tcast_kernel<<<tgrid, 256, 0, stream>>>(Wv, WvT, Dd, HKD);
  tcast_kernel<<<tgrid, 256, 0, stream>>>(Wo, WoT, HKD, Dd);

  // 3. Q/K/V projections (fold 1/sqrt(KD) into Q)
  const float qscale = 0.0625f;  // 1/sqrt(256)
  dim3 gproj(BS / 64, HKD / 64);
  gemm_qkv_kernel<<<gproj, 128, 0, stream>>>(Xq, WqT, bq, Qh, Dd, qscale, 0);
  gemm_qkv_kernel<<<gproj, 128, 0, stream>>>(Xk, WkT, bk, Kh, Dd, 1.0f, 0);
  gemm_qkv_kernel<<<gproj, 128, 0, stream>>>(Xv, WvT, bv, Vht, Dd, 1.0f, 1);

  // 4. Flash attention
  attn_kernel<<<dim3(S_LEN / 16, Hn, Bn), 32, 0, stream>>>(Qh, Kh, Vht, At);

  // 5. Output projection + bias + residual
  gemm_out_kernel<<<dim3(BS / 64, Dd / 64), 128, 0, stream>>>(At, WoT, bo, inq, out, HKD);
}
